// ChaosSNNSeizureDetector_28922309771720
// MI455X (gfx1250) — matrix operand order repaired and checked by reading, not compiled, not run on hardware
//
#include <hip/hip_runtime.h>
#include <hip/hip_bf16.h>

// ---------------------------------------------------------------------------
// ChaosSNNSeizureDetector for MI455X (gfx1250, wave32, WMMA)
//
// Stage 1: chaos modulator + threshold encoder  (parallel over 4096 sequences)
// Stage 2: persistent recurrent LIF reservoir   (16 WGs, f16 WMMA, LDS weights)
// Stage 3: readout (rate -> logit -> sigmoid)
// ---------------------------------------------------------------------------

typedef __attribute__((ext_vector_type(16))) _Float16 v16h;
typedef __attribute__((ext_vector_type(8)))  _Float16 v8h;
typedef __attribute__((ext_vector_type(8)))  float    v8f;
typedef __attribute__((ext_vector_type(4)))  float    v4f;

#define B_    64
#define C_    64
#define T_    2048
#define H_    512
#define G_    16          // persistent workgroups (must be co-resident; MI455X has >> 16 WGPs)
#define HS_   (H_ / G_)   // 32 output neurons per workgroup
#define WGT_  256         // 8 waves of 32

// exp(-DT/TAU_MEM) = exp(-0.05), exp(-DT/TAU_SYN) = exp(-0.1)
#define A_V 0.95122942450071402f
#define A_S 0.90483741803595957f

// ---------------------------------------------------------------------------
// Stage 1: chaos modulator + encoder. One thread per (b,c) sequence.
// Writes binary spikes as f16 in layout s_in[t][b][c] (c fastest) so the
// reservoir can load WMMA A-fragments with aligned b128 loads.
// ---------------------------------------------------------------------------
__global__ void chaos_encode_kernel(const float* __restrict__ x,
                                    const float* __restrict__ z0,
                                    _Float16* __restrict__ s_in)
{
    const int seq = blockIdx.x * blockDim.x + threadIdx.x;   // b*64 + c
    if (seq >= B_ * C_) return;

    const float* xp = x + (size_t)seq * T_;                  // x[b][c][:] contiguous in t
    float z    = z0[seq];
    float venc = 0.0f;

    for (int t0 = 0; t0 < T_; t0 += 4) {
        __builtin_prefetch(xp + t0 + 64, 0, 1);              // global_prefetch_b8
        v4f xv = *(const v4f*)(xp + t0);
#pragma unroll
        for (int j = 0; j < 4; ++j) {
            const float xt  = xv[j];
            const float sig = 3.5f * z * (1.0f - z) + 0.5f * xt;
            float zn = 0.5f * z + 0.25f * (1.0f + tanhf(sig));
            zn = fminf(fmaxf(zn, 0.0f), 1.0f);
            const float u = 0.5f * xt + 0.5f * (2.0f * zn - 1.0f);
            z = zn;
            venc = 0.9f * venc + u;
            const float s = (venc > 0.0f) ? 1.0f : 0.0f;
            venc -= s;
            s_in[(size_t)(t0 + j) * (B_ * C_) + seq] = (_Float16)s;
        }
    }
}

// ---------------------------------------------------------------------------
// Init: zero the spike ping buffer (s_prev(0) == 0) and the arrive counter.
// ---------------------------------------------------------------------------
__global__ void init_kernel(_Float16* __restrict__ S0, unsigned int* __restrict__ ctr)
{
    const int i = blockIdx.x * blockDim.x + threadIdx.x;
    if (i == 0) *ctr = 0u;
    if (i < B_ * H_) S0[i] = (_Float16)0.0f;
}

// ---------------------------------------------------------------------------
// WMMA A-fragment loader, 16-bit A 16x32 layout (CDNA5 ISA 7.12.2):
//   lanes 0-15  : M = lane,     halves 0-7 -> K=k0+0..7,  halves 8-15 -> K=k0+16..23
//   lanes 16-31 : M = lane-16,  halves 0-7 -> K=k0+8..15, halves 8-15 -> K=k0+24..31
// Two aligned 16B global loads per lane.
// ---------------------------------------------------------------------------
__device__ __forceinline__ v16h load_A(const _Float16* __restrict__ base, int ld,
                                       int m0, int k0)
{
    const int lane = threadIdx.x & 31;
    const int m  = m0 + (lane & 15);
    const int kb = k0 + ((lane & 16) ? 8 : 0);
    const _Float16* p = base + (size_t)m * ld + kb;
    v8h lo = *(const v8h*)(p);        // K = kb .. kb+7
    v8h hi = *(const v8h*)(p + 16);   // K = kb+16 .. kb+23
    v16h a;
#pragma unroll
    for (int i = 0; i < 8; ++i) { a[i] = lo[i]; a[i + 8] = hi[i]; }
    return a;
}

// ---------------------------------------------------------------------------
// Stage 2: persistent reservoir. 16 workgroups x 256 threads.
// WG g owns output neurons [g*32, g*32+32). Weights (transposed, scaled,
// f16) are staged once into LDS, pre-swizzled into B-fragment order
// [kt][nt][lane][16 halves] so each fragment is one aligned 32B ds read:
//   B layout (16-bit, 32x16): lanes 0-15: N=lane, K=kt*32 + (i&7) + 16*(i>=8);
//                             lanes 16-31: N=lane-16, K=kt*32 + 8 + (i&7) + 16*(i>=8).
// One 16x16 output tile per wave; K=64 (input) + K=512 (recurrent) = 18
// v_wmma_f32_16x16x32_f16 per wave per timestep, chained as two
// accumulators for XDL ILP. LIF state lives in registers using the C/D
// fragment mapping (lane L, vgpr r -> M = r + 8*(L>=16), N = L%16).
// Spike exchange: ping-pong global buffers + device-scope arrive counter
// (one barrier per step; ping-pong removes the WAR hazard).
// ---------------------------------------------------------------------------
__global__ void __launch_bounds__(WGT_, 1)
reservoir_kernel(const _Float16* __restrict__ s_in,
                 const float*    __restrict__ w_in,
                 const float*    __restrict__ w_rec,
                 const float*    __restrict__ bias,
                 _Float16*       __restrict__ S0,
                 _Float16*       __restrict__ S1,
                 float*          __restrict__ R,
                 unsigned int*   __restrict__ ctr)
{
    __shared__ _Float16 Brec[16 * 2 * 32 * 16];  // 32 KB : [kt16][nt2][lane32][i16]
    __shared__ _Float16 Bin [ 2 * 2 * 32 * 16];  //  4 KB : [kt2 ][nt2][lane32][i16]

    const int tid  = threadIdx.x;
    const int wg   = blockIdx.x;
    const int n0   = wg * HS_;
    const int wave = tid >> 5;
    const int lane = tid & 31;
    const int mt   = wave >> 1;   // batch tile 0..3
    const int nt   = wave & 1;    // neuron tile 0..1 within this WG's 32

    // ---- stage weights into LDS, transposed + pre-swizzled (done once) ----
    for (int idx = tid; idx < 16 * 2 * 32 * 16; idx += WGT_) {
        const int i   =  idx        & 15;
        const int ln  = (idx >> 4)  & 31;
        const int ntt = (idx >> 9)  &  1;
        const int kt  =  idx >> 10;
        const int n   = n0 + ntt * 16 + (ln & 15);
        const int k   = kt * 32 + (i & 7) + ((i & 8) ? 16 : 0) + ((ln & 16) ? 8 : 0);
        Brec[idx] = (_Float16)w_rec[(size_t)n * H_ + k];     // ALPHA_REC == 1
    }
    for (int idx = tid; idx < 2 * 2 * 32 * 16; idx += WGT_) {
        const int i   =  idx        & 15;
        const int ln  = (idx >> 4)  & 31;
        const int ntt = (idx >> 9)  &  1;
        const int kt  =  idx >> 10;
        const int n   = n0 + ntt * 16 + (ln & 15);
        const int k   = kt * 32 + (i & 7) + ((i & 8) ? 16 : 0) + ((ln & 16) ? 8 : 0);
        Bin[idx] = (_Float16)w_in[(size_t)n * C_ + k];
    }
    __syncthreads();

    // per-lane LIF state, C/D fragment mapping of this wave's tile
    const int   m_base = mt * 16 + ((lane & 16) ? 8 : 0);
    const int   n_glob = n0 + nt * 16 + (lane & 15);
    const float bn     = bias[n_glob];

    float vmem[8], Isyn[8], cnt[8];
#pragma unroll
    for (int r = 0; r < 8; ++r) { vmem[r] = 0.0f; Isyn[r] = 0.0f; cnt[r] = 0.0f; }

    for (int t = 0; t < T_; ++t) {
        const _Float16* Sprev = (t & 1) ? S1 : S0;
        _Float16*       Snext = (t & 1) ? S0 : S1;

        v8f acc0 = {0.f,0.f,0.f,0.f,0.f,0.f,0.f,0.f};
        v8f acc1 = {0.f,0.f,0.f,0.f,0.f,0.f,0.f,0.f};

        // input projection: s_t (64x64) @ w_in^T,  K = 64
        const _Float16* At = s_in + (size_t)t * (B_ * C_);
#pragma unroll
        for (int kt = 0; kt < 2; ++kt) {
            v16h a = load_A(At, C_, mt * 16, kt * 32);
            v16h b = *(const v16h*)&Bin[(((size_t)kt * 2 + nt) * 32 + lane) * 16];
            if (kt & 1)
                acc1 = __builtin_amdgcn_wmma_f32_16x16x32_f16(false, a, false, b, (short)0, acc1, false, false);
            else
                acc0 = __builtin_amdgcn_wmma_f32_16x16x32_f16(false, a, false, b, (short)0, acc0, false, false);
        }
        // recurrent: s_prev (64x512) @ w_rec^T,  K = 512
#pragma unroll
        for (int kt = 0; kt < 16; ++kt) {
            v16h a = load_A(Sprev, H_, mt * 16, kt * 32);
            v16h b = *(const v16h*)&Brec[(((size_t)kt * 2 + nt) * 32 + lane) * 16];
            if (kt & 1)
                acc1 = __builtin_amdgcn_wmma_f32_16x16x32_f16(false, a, false, b, (short)0, acc1, false, false);
            else
                acc0 = __builtin_amdgcn_wmma_f32_16x16x32_f16(false, a, false, b, (short)0, acc0, false, false);
        }

        // LIF update + spike, all in registers (D-fragment element mapping)
#pragma unroll
        for (int r = 0; r < 8; ++r) {
            const float In = A_S * Isyn[r] + (acc0[r] + acc1[r]);
            Isyn[r] = In;
            const float vv = A_V * vmem[r] + In + bn;
            const float sp = (vv > 1.0f) ? 1.0f : 0.0f;
            vmem[r] = vv - sp;
            cnt[r] += sp;
            Snext[(size_t)(m_base + r) * H_ + n_glob] = (_Float16)sp;
        }

        // device-wide step barrier (monotonic arrive counter)
        __threadfence();
        __syncthreads();
        if (tid == 0) {
            atomicAdd(ctr, 1u);
            const unsigned int target = (unsigned int)G_ * (unsigned int)(t + 1);
            while (__hip_atomic_load(ctr, __ATOMIC_RELAXED, __HIP_MEMORY_SCOPE_AGENT) < target) {
                __builtin_amdgcn_s_sleep(1);
            }
        }
        __syncthreads();
        __threadfence();
    }

    // spike counts -> rates buffer (readout divides by T)
#pragma unroll
    for (int r = 0; r < 8; ++r)
        R[(size_t)(m_base + r) * H_ + n_glob] = cnt[r];
}

// ---------------------------------------------------------------------------
// Stage 3: readout. 64 threads, one per batch element.
// ---------------------------------------------------------------------------
__global__ void readout_kernel(const float* __restrict__ R,
                               const float* __restrict__ w_out,
                               const float* __restrict__ b_out,
                               float* __restrict__ out)
{
    const int b = blockIdx.x * blockDim.x + threadIdx.x;
    if (b >= B_) return;
    float acc = 0.0f;
    for (int h = 0; h < H_; ++h)
        acc += R[(size_t)b * H_ + h] * w_out[h];
    const float logit = acc * (1.0f / (float)T_) + b_out[0];
    out[b] = 1.0f / (1.0f + __expf(-logit));
}

// ---------------------------------------------------------------------------
// Launch. Inputs (setup_inputs order): x, z0, w_in, w_rec_raw, bias, w_out, b_out.
// Output: 64 float32 sigmoid probabilities.
// Workspace: s_in f16 (16 MB) | S0 | S1 | R | ctr  (~17 MB total)
// ---------------------------------------------------------------------------
extern "C" void kernel_launch(void* const* d_in, const int* in_sizes, int n_in,
                              void* d_out, int out_size, void* d_ws, size_t ws_size,
                              hipStream_t stream)
{
    (void)in_sizes; (void)n_in; (void)out_size; (void)ws_size;

    const float* x     = (const float*)d_in[0];
    const float* z0    = (const float*)d_in[1];
    const float* w_in  = (const float*)d_in[2];
    const float* w_rec = (const float*)d_in[3];
    const float* bias  = (const float*)d_in[4];
    const float* w_out = (const float*)d_in[5];
    const float* b_out = (const float*)d_in[6];
    float* out = (float*)d_out;

    char* ws = (char*)d_ws;
    _Float16* s_in = (_Float16*)ws;  ws += (size_t)T_ * B_ * C_ * sizeof(_Float16);
    _Float16* S0   = (_Float16*)ws;  ws += (size_t)B_ * H_ * sizeof(_Float16);
    _Float16* S1   = (_Float16*)ws;  ws += (size_t)B_ * H_ * sizeof(_Float16);
    float*    R    = (float*)ws;     ws += (size_t)B_ * H_ * sizeof(float);
    unsigned int* ctr = (unsigned int*)ws;

    hipLaunchKernelGGL(init_kernel, dim3((B_ * H_ + 255) / 256), dim3(256), 0, stream,
                       S0, ctr);
    hipLaunchKernelGGL(chaos_encode_kernel, dim3((B_ * C_ + 255) / 256), dim3(256), 0, stream,
                       x, z0, s_in);
    hipLaunchKernelGGL(reservoir_kernel, dim3(G_), dim3(WGT_), 0, stream,
                       s_in, w_in, w_rec, bias, S0, S1, R, ctr);
    hipLaunchKernelGGL(readout_kernel, dim3(1), dim3(64), 0, stream,
                       R, w_out, b_out, out);
}
